// RetrievalLoss_14121852469881
// MI455X (gfx1250) — matrix operand order repaired and checked by reading, not compiled, not run on hardware
//
#include <hip/hip_runtime.h>

typedef __bf16  v16bf __attribute__((ext_vector_type(16)));
typedef float   v8f   __attribute__((ext_vector_type(8)));

#define N_ROWS 8192
#define DIMK   512
#define NCHUNK 8
#define CHUNK_COLS 1024   // 64 column tiles of 16
#define DELTA 1.0f

union BF16x16 { v16bf v; uint4 u[2]; };

static __device__ __forceinline__ unsigned short bf_bits(__bf16 b) {
  union { __bf16 b; unsigned short u; } x; x.b = b; return x.u;
}

// ---------------------------------------------------------------------------
// Kernel 1: split fp32 -> bf16 hi/lo, and compute row L2 norms (fp32).
// One wave per row, 8 waves (256 threads) per block.
// ---------------------------------------------------------------------------
__global__ __launch_bounds__(256) void prep_kernel(const float* __restrict__ q,
                                                   unsigned short* __restrict__ qhi,
                                                   unsigned short* __restrict__ qlo,
                                                   float* __restrict__ l2q) {
  const int row  = blockIdx.x * 8 + (threadIdx.x >> 5);
  const int lane = threadIdx.x & 31;
  const float4* qr = (const float4*)(q + (size_t)row * DIMK);
  ushort4* hr = (ushort4*)(qhi + (size_t)row * DIMK);
  ushort4* lr = (ushort4*)(qlo + (size_t)row * DIMK);

  float s = 0.0f;
#pragma unroll
  for (int t = 0; t < 4; ++t) {
    float4 v = qr[lane + t * 32];
    s = fmaf(v.x, v.x, s); s = fmaf(v.y, v.y, s);
    s = fmaf(v.z, v.z, s); s = fmaf(v.w, v.w, s);
    __bf16 hx = (__bf16)v.x, hy = (__bf16)v.y, hz = (__bf16)v.z, hw = (__bf16)v.w;
    __bf16 lx = (__bf16)(v.x - (float)hx);
    __bf16 ly = (__bf16)(v.y - (float)hy);
    __bf16 lz = (__bf16)(v.z - (float)hz);
    __bf16 lw = (__bf16)(v.w - (float)hw);
    ushort4 h4; h4.x = bf_bits(hx); h4.y = bf_bits(hy); h4.z = bf_bits(hz); h4.w = bf_bits(hw);
    ushort4 l4; l4.x = bf_bits(lx); l4.y = bf_bits(ly); l4.z = bf_bits(lz); l4.w = bf_bits(lw);
    hr[lane + t * 32] = h4;
    lr[lane + t * 32] = l4;
  }
#pragma unroll
  for (int m = 16; m >= 1; m >>= 1) s += __shfl_xor(s, m, 32);
  if (lane == 0) l2q[row] = s;
}

// ---------------------------------------------------------------------------
// Kernel 2: fused bf16x3 WMMA GEMM + masked hardest-pos/neg argmax.
// Each wave owns a 16-row tile; grid.y picks a 1024-column chunk.
// 4 waves per block (4 row tiles). 4 column tiles per accumulation group.
// B/A base pointers hoisted out of the K loop so per-K addressing folds into
// immediate offsets (no per-iteration 64-bit shifts/adds).
// ---------------------------------------------------------------------------
__global__ __launch_bounds__(128) void mine_kernel(const unsigned short* __restrict__ qhi_,
                                                   const unsigned short* __restrict__ qlo_,
                                                   const float* __restrict__ l2q,
                                                   const int* __restrict__ tgt,
                                                   float* __restrict__ pval, int* __restrict__ pidx,
                                                   float* __restrict__ nval, int* __restrict__ nidx) {
  const __bf16* qhi = (const __bf16*)qhi_;
  const __bf16* qlo = (const __bf16*)qlo_;

  const int lane  = threadIdx.x & 31;
  const int wave  = threadIdx.x >> 5;
  const int rtile = blockIdx.x * 4 + wave;
  const int chunk = blockIdx.y;
  const int rbase = rtile * 16;
  const int cbase = chunk * CHUNK_COLS;
  const int n = lane & 15;
  const int h = lane >> 4;

  // Per-lane row constants: this lane's C rows are M = r + 8*h, r = 0..7.
  float l2r[8]; int trow[8];
#pragma unroll
  for (int r = 0; r < 8; ++r) {
    l2r[r]  = l2q[rbase + r + 8 * h];
    trow[r] = tgt[rbase + r + 8 * h];
  }

  float pv[8], nv[8]; int pi[8], ni[8];
#pragma unroll
  for (int r = 0; r < 8; ++r) { pv[r] = -3.4e38f; nv[r] = -3.4e38f; pi[r] = 0; ni[r] = 0; }

  // A fragment base pointers (fixed for the whole kernel):
  // lanes 0-15 and 16-31 both hold rows M=0..15; K runs [h*16,h*16+16).
  const size_t aoff = (size_t)(rbase + n) * DIMK + h * 16;
  const __bf16* ahP = qhi + aoff;
  const __bf16* alP = qlo + aoff;

  for (int cg = 0; cg < CHUNK_COLS / 64; ++cg) {   // 16 groups of 4 column tiles
    const int c0 = cbase + cg * 64;

    // Hoisted B column base pointers: lane holds column N=n of each tile,
    // 16 contiguous K starting at h*16.
    const __bf16* bh[4];
    const __bf16* bl[4];
#pragma unroll
    for (int ct = 0; ct < 4; ++ct) {
      const size_t o = (size_t)(c0 + ct * 16 + n) * DIMK + h * 16;
      bh[ct] = qhi + o;
      bl[ct] = qlo + o;
    }

    v8f acc[4] = {};

#pragma unroll 4
    for (int k0 = 0; k0 < DIMK; k0 += 32) {
      BF16x16 ahi, alo;
      ahi.u[0] = *(const uint4*)(ahP + k0);
      ahi.u[1] = *(const uint4*)(ahP + k0 + 8);
      alo.u[0] = *(const uint4*)(alP + k0);
      alo.u[1] = *(const uint4*)(alP + k0 + 8);

#pragma unroll
      for (int ct = 0; ct < 4; ++ct) {
        BF16x16 bhi, blo;
        bhi.u[0] = ((const uint4*)(bh[ct] + k0))[0];
        bhi.u[1] = ((const uint4*)(bh[ct] + k0))[1];
        blo.u[0] = ((const uint4*)(bl[ct] + k0))[0];
        blo.u[1] = ((const uint4*)(bl[ct] + k0))[1];
        acc[ct] = __builtin_amdgcn_wmma_f32_16x16x32_bf16(false, ahi.v, false, bhi.v, (short)0, acc[ct], false, false);
        acc[ct] = __builtin_amdgcn_wmma_f32_16x16x32_bf16(false, ahi.v, false, blo.v, (short)0, acc[ct], false, false);
        acc[ct] = __builtin_amdgcn_wmma_f32_16x16x32_bf16(false, alo.v, false, bhi.v, (short)0, acc[ct], false, false);
      }
    }

    // Epilogue: d[i,j] = -2*acc + 2*l2q[i]; mask-multiply argmax (masked -> 0.0)
#pragma unroll
    for (int ct = 0; ct < 4; ++ct) {
      const int col = c0 + ct * 16 + n;
      const int tc = tgt[col];
#pragma unroll
      for (int r = 0; r < 8; ++r) {
        const float dv = fmaf(-2.0f, acc[ct][r], l2r[r] + l2r[r]);
        const bool same = (trow[r] == tc);
        const float pc = same ? dv : 0.0f;
        const float nc = same ? 0.0f : dv;
        if (pc > pv[r]) { pv[r] = pc; pi[r] = col; }
        if (nc > nv[r]) { nv[r] = nc; ni[r] = col; }
      }
    }
  }

  // Butterfly reduce across the 16 lanes holding each row (xor 1,2,4,8 stays in half-wave).
#pragma unroll
  for (int m = 1; m <= 8; m <<= 1) {
#pragma unroll
    for (int r = 0; r < 8; ++r) {
      float ov = __shfl_xor(pv[r], m, 32); int oi = __shfl_xor(pi[r], m, 32);
      if (ov > pv[r] || (ov == pv[r] && oi < pi[r])) { pv[r] = ov; pi[r] = oi; }
      ov = __shfl_xor(nv[r], m, 32); oi = __shfl_xor(ni[r], m, 32);
      if (ov > nv[r] || (ov == nv[r] && oi < ni[r])) { nv[r] = ov; ni[r] = oi; }
    }
  }

  if (n == 0) {  // lanes 0 (rows 0..7) and 16 (rows 8..15)
#pragma unroll
    for (int r = 0; r < 8; ++r) {
      const int row = rbase + r + 8 * h;
      pval[chunk * N_ROWS + row] = pv[r];
      pidx[chunk * N_ROWS + row] = pi[r];
      nval[chunk * N_ROWS + row] = nv[r];
      nidx[chunk * N_ROWS + row] = ni[r];
    }
  }
}

// ---------------------------------------------------------------------------
// Kernel 3: resolve chunk partials, compute per-row triplet loss (fp32 exact).
// One wave per row.
// ---------------------------------------------------------------------------
__global__ __launch_bounds__(256) void loss_kernel(const float* __restrict__ q,
                                                   const float* __restrict__ pval, const int* __restrict__ pidx,
                                                   const float* __restrict__ nval, const int* __restrict__ nidx,
                                                   float* __restrict__ lossOut) {
  const int row  = blockIdx.x * 8 + (threadIdx.x >> 5);
  const int lane = threadIdx.x & 31;

  int bp = 0, bn = 0;
  if (lane == 0) {
    float bvp = -3.4e38f, bvn = -3.4e38f; int bip = 0, bin = 0;
    for (int c = 0; c < NCHUNK; ++c) {
      float v = pval[c * N_ROWS + row]; int i = pidx[c * N_ROWS + row];
      if (v > bvp || (v == bvp && i < bip)) { bvp = v; bip = i; }
      v = nval[c * N_ROWS + row]; i = nidx[c * N_ROWS + row];
      if (v > bvn || (v == bvn && i < bin)) { bvn = v; bin = i; }
    }
    bp = bip; bn = bin;
  }
  bp = __shfl(bp, 0, 32);
  bn = __shfl(bn, 0, 32);

  const float4* qi = (const float4*)(q + (size_t)row * DIMK);
  const float4* qp = (const float4*)(q + (size_t)bp * DIMK);
  const float4* qn = (const float4*)(q + (size_t)bn * DIMK);
  float sp = 0.0f, sn = 0.0f;
#pragma unroll
  for (int t = 0; t < 4; ++t) {
    float4 a = qi[lane + t * 32], p = qp[lane + t * 32], g = qn[lane + t * 32];
    float d;
    d = a.x - p.x; sp = fmaf(d, d, sp);  d = a.y - p.y; sp = fmaf(d, d, sp);
    d = a.z - p.z; sp = fmaf(d, d, sp);  d = a.w - p.w; sp = fmaf(d, d, sp);
    d = a.x - g.x; sn = fmaf(d, d, sn);  d = a.y - g.y; sn = fmaf(d, d, sn);
    d = a.z - g.z; sn = fmaf(d, d, sn);  d = a.w - g.w; sn = fmaf(d, d, sn);
  }
#pragma unroll
  for (int m = 16; m >= 1; m >>= 1) { sp += __shfl_xor(sp, m, 32); sn += __shfl_xor(sn, m, 32); }
  if (lane == 0) lossOut[row] = fmaxf(0.0f, DELTA - sp + sn);
}

// ---------------------------------------------------------------------------
// Kernel 4: deterministic mean over 8192 losses (single block, fixed order).
// ---------------------------------------------------------------------------
__global__ __launch_bounds__(256) void mean_kernel(const float* __restrict__ loss,
                                                   float* __restrict__ out) {
  __shared__ float sm[256];
  const int t = threadIdx.x;
  float s = 0.0f;
  for (int i = 0; i < N_ROWS / 256; ++i) s += loss[t + i * 256];
  sm[t] = s;
  __syncthreads();
  for (int off = 128; off > 0; off >>= 1) {
    if (t < off) sm[t] += sm[t + off];
    __syncthreads();
  }
  if (t == 0) out[0] = sm[0] * (1.0f / (float)N_ROWS);
}

// ---------------------------------------------------------------------------
extern "C" void kernel_launch(void* const* d_in, const int* in_sizes, int n_in,
                              void* d_out, int out_size, void* d_ws, size_t ws_size,
                              hipStream_t stream) {
  const float* q   = (const float*)d_in[0];
  const int*   tgt = (const int*)d_in[1];
  float*       out = (float*)d_out;

  char* ws = (char*)d_ws;
  size_t off = 0;
  unsigned short* qhi = (unsigned short*)(ws + off); off += (size_t)N_ROWS * DIMK * 2;
  unsigned short* qlo = (unsigned short*)(ws + off); off += (size_t)N_ROWS * DIMK * 2;
  float* l2q  = (float*)(ws + off); off += (size_t)N_ROWS * 4;
  float* pval = (float*)(ws + off); off += (size_t)NCHUNK * N_ROWS * 4;
  int*   pidx = (int*)  (ws + off); off += (size_t)NCHUNK * N_ROWS * 4;
  float* nval = (float*)(ws + off); off += (size_t)NCHUNK * N_ROWS * 4;
  int*   nidx = (int*)  (ws + off); off += (size_t)NCHUNK * N_ROWS * 4;
  float* loss = (float*)(ws + off); off += (size_t)N_ROWS * 4;

  prep_kernel<<<N_ROWS / 8, 256, 0, stream>>>(q, qhi, qlo, l2q);

  dim3 gridB(N_ROWS / 16 / 4, NCHUNK);
  mine_kernel<<<gridB, 128, 0, stream>>>(qhi, qlo, l2q, tgt, pval, pidx, nval, nidx);

  loss_kernel<<<N_ROWS / 8, 256, 0, stream>>>(q, pval, pidx, nval, nidx, loss);

  mean_kernel<<<1, 256, 0, stream>>>(loss, out);
}
